// Tokenizer_31988916420619
// MI455X (gfx1250) — matrix operand order repaired and checked by reading, not compiled, not run on hardware
//
#include <hip/hip_runtime.h>
#include <hip/hip_bf16.h>

// ---------------------------------------------------------------------------
// Tokenizer on MI455X (gfx1250): bf16 WMMA (v_wmma_f32_16x16x32_bf16) for all
// GEMMs, fp32 accumulation, async global->LDS staging (ASYNCcnt) for shared
// B-panels. Wave32, 16x16x32 tiles, contiguous-K fragment layouts.
// ---------------------------------------------------------------------------

#define NB     32
#define CIN    512
#define HWD    3136      // 56*56 = 49*64 = 98*32
#define LTOK   64
#define CT     512
#define GROUPS 16
#define POSD   49
#define KCAT   576       // 512 + 49 padded to 576 (18 chunks of 32)
#define KSPLIT 7         // HW split: 7 blocks * 14 chunks * 32 cols = 3136

typedef __attribute__((ext_vector_type(16))) __bf16 v16bf;
typedef __attribute__((ext_vector_type(8)))  float  v8f;

#define WMMA_BF16(A, B, C) \
  __builtin_amdgcn_wmma_f32_16x16x32_bf16(false, (A), false, (B), (short)0, (C), false, false)

// float -> bf16 (round to nearest even), returned as raw u16
__device__ __forceinline__ unsigned short f2bf(float f) {
  unsigned int u = __float_as_uint(f);
  u += 0x7FFFu + ((u >> 16) & 1u);
  return (unsigned short)(u >> 16);
}

// Async 16-byte global -> LDS copy (tracked by ASYNCcnt). Generic LDS pointer
// low 32 bits are the LDS address (flat aperture rule, ISA 10.2).
__device__ __forceinline__ void async_b128(const void* gsrc, void* lds_dst) {
  unsigned loff = (unsigned)(unsigned long long)lds_dst;
  unsigned long long g = (unsigned long long)gsrc;
  asm volatile("global_load_async_to_lds_b128 %0, %1, off" ::"v"(loff), "v"(g)
               : "memory");
}
__device__ __forceinline__ void wait_async0() {
  asm volatile("s_wait_asynccnt 0x0" ::: "memory");
}

// ---- fragment builders (contiguous K order, k = 16*(lane/16) + e; wave32) ------------------
// A (16x32 bf16): lanes 0-15 row M=lane: element e == K=0..15;
//                 lanes 16-31 same rows: K=16..31.
__device__ __forceinline__ int a_kidx(int e, int lane) {
  return e + ((lane < 16) ? 0 : 16);
}

// A-frag from fp32 row-major [m][k] (ld floats)
__device__ __forceinline__ v16bf load_a_f32(const float* __restrict__ p, int m0, int k0, int ld) {
  int t = threadIdx.x & 31;
  const float* row = p + (size_t)(m0 + (t & 15)) * ld + k0;
  union { v16bf v; unsigned short s[16]; } U;
#pragma unroll
  for (int e = 0; e < 16; ++e) U.s[e] = f2bf(row[a_kidx(e, t)]);
  return U.v;
}

// A-frag from bf16(u16) row-major [m][k] in LDS
__device__ __forceinline__ v16bf load_a_bf_lds(const unsigned short* p, int m0, int k0, int ld) {
  int t = threadIdx.x & 31;
  const unsigned short* row = p + (size_t)(m0 + (t & 15)) * ld + k0;
  union { v16bf v; unsigned short s[16]; } U;
#pragma unroll
  for (int e = 0; e < 16; ++e) U.s[e] = row[a_kidx(e, t)];
  return U.v;
}

// A-frag from fp32 w_tok (row stride 561) with K zero-padding past 561
__device__ __forceinline__ v16bf load_a_wtok(const float* __restrict__ p, int m0, int k0) {
  int t = threadIdx.x & 31;
  const float* row = p + (size_t)(m0 + (t & 15)) * (CT + POSD);
  union { v16bf v; unsigned short s[16]; } U;
#pragma unroll
  for (int e = 0; e < 16; ++e) {
    int k = k0 + a_kidx(e, t);
    U.s[e] = (k < CT + POSD) ? f2bf(row[k]) : (unsigned short)0;
  }
  return U.v;
}

// B (32x16 bf16): lane = column; lanes 0-15 hold K=0..15 (element e == K),
//                 lanes 16-31 hold K=16..31.
// B-frag from fp32 [k][n] row-major (ld floats): fixed col per lane; per
// element the 16-lane half-wave reads 64B contiguous -> coalesced.
__device__ __forceinline__ v16bf load_b_f32(const float* __restrict__ p, int k0, int n0, int ld) {
  int t = threadIdx.x & 31;
  int c = n0 + (t & 15);
  int kb = (t < 16) ? 0 : 16;
  union { v16bf v; unsigned short s[16]; } U;
#pragma unroll
  for (int e = 0; e < 16; ++e) U.s[e] = f2bf(p[(size_t)(k0 + kb + e) * ld + c]);
  return U.v;
}

// B-frag from bf16 [k][n] row-major (u16, ld)
__device__ __forceinline__ v16bf load_b_bf(const unsigned short* __restrict__ p, int k0, int n0, int ld) {
  int t = threadIdx.x & 31;
  int c = n0 + (t & 15);
  int kb = (t < 16) ? 0 : 16;
  union { v16bf v; unsigned short s[16]; } U;
#pragma unroll
  for (int e = 0; e < 16; ++e) U.s[e] = p[(size_t)(k0 + kb + e) * ld + c];
  return U.v;
}

// B-frag from bf16 stored transposed [col][K=32] in LDS: 32B contiguous per
// lane -> two ds_load_b128.
__device__ __forceinline__ v16bf load_b_lds32(const unsigned short* p, int c0) {
  int t = threadIdx.x & 31;
  int c = c0 + (t & 15);
  int kb = (t < 16) ? 0 : 16;
  union { v16bf v; uint4 q[2]; } U;
  const uint4* s = (const uint4*)(p + c * 32 + kb);
  U.q[0] = s[0];
  U.q[1] = s[1];
  return U.v;
}

// B-frag from bf16 stored transposed [col][K] in global (contiguous K)
__device__ __forceinline__ v16bf load_b_bfT(const unsigned short* __restrict__ p, int k0, int c0,
                                            size_t ldK) {
  int t = threadIdx.x & 31;
  int c = c0 + (t & 15);
  int kb = (t < 16) ? 0 : 16;
  union { v16bf v; uint4 q[2]; } U;
  const uint4* src = (const uint4*)(p + (size_t)c * ldK + k0 + kb);
  U.q[0] = src[0];
  U.q[1] = src[1];
  return U.v;
}

// ---------------------------------------------------------------------------
// K4 (listed first so disasm snippet shows the async path): fused grouped-conv
// val + token aggregation. grid(7, 32), 512 thr. Wave w == group w: WMMA val
// slice (32x32), re-layout via wave-private LDS slice (DS in-order per wave),
// then WMMA aggregation against the tc panel, which is double-buffer
// async-copied global->LDS (global_load_async_to_lds_b128 + s_wait_asynccnt)
// and shared by all 16 waves.  Deterministic K-split partials.
// ---------------------------------------------------------------------------
__global__ __launch_bounds__(512) void tok_val_agg(const float* __restrict__ feat,
                                                   const float* __restrict__ w_val,
                                                   const unsigned short* __restrict__ tcb,
                                                   float* __restrict__ part) {
  int ks = blockIdx.x;  // 0..6
  int n = blockIdx.y;
  int w = threadIdx.x >> 5;  // wave == group == CT row-block
  __shared__ unsigned short valS[CT * 32];    // 32 KB: wave-private 32x32 slices
  __shared__ unsigned short tcS[2][64 * 32];  // 2 x 4 KB: [l][i] bf16 panel
  const float* fg = feat + (size_t)n * CIN * HWD + (size_t)w * 32 * HWD;
  const unsigned short* tcn = tcb + (size_t)n * LTOK * HWD;
  unsigned short* myval = valS + w * 32 * 32;
  int t = threadIdx.x & 31;
  int hi = (t < 16) ? 0 : 8, cl = t & 15;

  // w_val A-fragments for this group (reused across all chunks)
  v16bf wa0 = load_a_f32(w_val + (size_t)w * 32 * 32, 0, 0, 32);
  v16bf wa1 = load_a_f32(w_val + (size_t)w * 32 * 32, 16, 0, 32);

  // preload tc chunk 0 (waves 0..7 issue; 256 lanes x 16B = 4 KB panel)
  if (threadIdx.x < 256) {
    int l = threadIdx.x >> 2, sg = threadIdx.x & 3;
    async_b128(tcn + (size_t)l * HWD + (ks * 14) * 32 + sg * 8, &tcS[0][l * 32 + sg * 8]);
  }

  v8f acc[8] = {};  // 2 (M) x 4 (N) tiles of tokens rows [w*32, w*32+32)
  for (int cc = 0; cc < 14; ++cc) {
    int i0 = (ks * 14 + cc) * 32;
    wait_async0();
    __syncthreads();  // publish tcS[cc&1] to all waves; retire prior reads
    if (cc + 1 < 14 && threadIdx.x < 256) {
      int l = threadIdx.x >> 2, sg = threadIdx.x & 3;
      async_b128(tcn + (size_t)l * HWD + (i0 + 32) + sg * 8,
                 &tcS[(cc + 1) & 1][l * 32 + sg * 8]);
    }
    // ---- val slice: (32x32 w_val) x (32x32 feature chunk), K=32 ----
#pragma unroll
    for (int nt = 0; nt < 2; ++nt) {
      v16bf b = load_b_f32(fg, 0, i0 + nt * 16, HWD);
      v8f c0 = {}, c1 = {};
      c0 = WMMA_BF16(wa0, b, c0);
      c1 = WMMA_BF16(wa1, b, c1);
#pragma unroll
      for (int r = 0; r < 8; ++r) {
        myval[(r + hi) * 32 + nt * 16 + cl] = f2bf(c0[r]);
        myval[(16 + r + hi) * 32 + nt * 16 + cl] = f2bf(c1[r]);
      }
    }
    // ---- aggregation: tokens += val(32xK=32) x tc(K=32 x 64) ----
    v16bf a0 = load_a_bf_lds(myval, 0, 0, 32);
    v16bf a1 = load_a_bf_lds(myval, 16, 0, 32);
    const unsigned short* bbuf = tcS[cc & 1];
#pragma unroll
    for (int nt = 0; nt < 4; ++nt) {
      v16bf b = load_b_lds32(bbuf, nt * 16);
      acc[nt] = WMMA_BF16(a0, b, acc[nt]);
      acc[4 + nt] = WMMA_BF16(a1, b, acc[4 + nt]);
    }
  }
  float* pp = part + ((size_t)n * KSPLIT + ks) * CT * 64;
#pragma unroll
  for (int mt = 0; mt < 2; ++mt)
#pragma unroll
    for (int nt = 0; nt < 4; ++nt)
#pragma unroll
      for (int r = 0; r < 8; ++r) {
        int m = w * 32 + mt * 16 + r + hi;
        pp[(size_t)m * 64 + nt * 16 + cl] = acc[mt * 4 + nt][r];
      }
}

// ---------------------------------------------------------------------------
// K1: logits = (w_tc @ feature[n]) scaled.  grid(49, 32), 128 thr (4 waves).
// Each wave owns one M-tile x 4 N-tiles: A-frag built once per K-step and
// reused across 4 WMMAs. LDS-free; feature B-frags 64B-coalesced.
// ---------------------------------------------------------------------------
__global__ __launch_bounds__(128) void tok_tc_gemm(const float* __restrict__ feat,
                                                   const float* __restrict__ w_tc,
                                                   const float* __restrict__ b_tc,
                                                   float* __restrict__ logits) {
  int n = blockIdx.y;
  int i0 = blockIdx.x * 64;
  int w = threadIdx.x >> 5;
  int mi = w * 16;
  const float* fn = feat + (size_t)n * CIN * HWD;
  v8f acc[4] = {};
  for (int k0 = 0; k0 < CIN; k0 += 32) {
    if (k0 + 32 < CIN)
      __builtin_prefetch(fn + (size_t)(k0 + 32) * HWD + i0 + (threadIdx.x & 31), 0, 1);
    v16bf a = load_a_f32(w_tc, mi, k0, CIN);
#pragma unroll
    for (int nt = 0; nt < 4; ++nt) {
      v16bf b = load_b_f32(fn, k0, i0 + nt * 16, HWD);
      acc[nt] = WMMA_BF16(a, b, acc[nt]);
    }
  }
  const float scale = 0.04419417382415922f;  // 1/sqrt(512)
  int t = threadIdx.x & 31;
  int hi = (t < 16) ? 0 : 8, cl = t & 15;
#pragma unroll
  for (int nt = 0; nt < 4; ++nt) {
    int col = i0 + nt * 16 + cl;
#pragma unroll
    for (int r = 0; r < 8; ++r) {
      int l = mi + r + hi;
      logits[((size_t)n * LTOK + l) * HWD + col] = (acc[nt][r] + b_tc[l]) * scale;
    }
  }
}

// ---------------------------------------------------------------------------
// K2: softmax over HW per (n,l). Writes bf16 tc in [n][l][i] layout + stats.
// ---------------------------------------------------------------------------
__global__ __launch_bounds__(256) void tok_softmax(const float* __restrict__ logits,
                                                   unsigned short* __restrict__ tcb,
                                                   float* __restrict__ stats) {
  int nl = blockIdx.x;
  const float* row = logits + (size_t)nl * HWD;
  __shared__ float red[256];
  int t = threadIdx.x;
  float mx = -1e30f;
  for (int i = t; i < HWD; i += 256) mx = fmaxf(mx, row[i]);
  red[t] = mx;
  __syncthreads();
  for (int s = 128; s > 0; s >>= 1) {
    if (t < s) red[t] = fmaxf(red[t], red[t + s]);
    __syncthreads();
  }
  mx = red[0];
  __syncthreads();
  float sum = 0.f;
  for (int i = t; i < HWD; i += 256) sum += __expf(row[i] - mx);
  red[t] = sum;
  __syncthreads();
  for (int s = 128; s > 0; s >>= 1) {
    if (t < s) red[t] += red[t + s];
    __syncthreads();
  }
  float inv = 1.f / red[0];
  unsigned short* out = tcb + (size_t)nl * HWD;
  for (int i = t; i < HWD; i += 256) out[i] = f2bf(__expf(row[i] - mx) * inv);
  if (t == 0) { stats[2 * nl] = mx; stats[2 * nl + 1] = inv; }
}

// ---------------------------------------------------------------------------
// K3: PosEncoder. torch .view quirk: image (n,l') pixel p maps to
// tc[n, i = l'*49 + p/64, col = p%64]. grid(32*64), 64 thr.
// ---------------------------------------------------------------------------
__global__ __launch_bounds__(64) void tok_pos(const float* __restrict__ logits,
                                              const float* __restrict__ stats,
                                              const float* __restrict__ w_ds3,
                                              const float* __restrict__ b_ds3,
                                              const float* __restrict__ w_ds1,
                                              const float* __restrict__ b_ds1,
                                              const float* __restrict__ w_pos,
                                              const float* __restrict__ b_pos,
                                              float* __restrict__ pos) {
  int n = blockIdx.x >> 6, lp = blockIdx.x & 63;
  __shared__ float g[16][16];  // 14x14 grid, pad=1
  __shared__ float pc[49];
  int t = threadIdx.x;
  for (int idx = t; idx < 256; idx += 64) ((float*)g)[idx] = 0.f;
  __syncthreads();
  for (int idx = t; idx < 196; idx += 64) {
    int y = idx / 14, x = idx % 14;
    int p = 224 * y + 4 * x;  // nearest sample (4y, 4x)
    int i = lp * POSD + (p >> 6);
    int col = p & 63;
    int nl = n * 64 + col;
    float v = __expf(logits[(size_t)nl * HWD + i] - stats[2 * nl]) * stats[2 * nl + 1];
    g[y + 1][x + 1] = v;
  }
  __syncthreads();
  if (t < 49) {
    int oy = t / 7, ox = t % 7;
    float s = 0.f;
#pragma unroll
    for (int dy = 0; dy < 3; ++dy)
#pragma unroll
      for (int dx = 0; dx < 3; ++dx) s += w_ds3[dy * 3 + dx] * g[2 * oy + dy][2 * ox + dx];
    s += b_ds3[0];
    pc[t] = s * w_ds1[0] + b_ds1[0];
  }
  __syncthreads();
  if (t < 49) {
    float s = b_pos[t];
#pragma unroll 7
    for (int c = 0; c < 49; ++c) s += w_pos[t * 49 + c] * pc[c];
    pos[((size_t)n * POSD + t) * LTOK + lp] = s;
  }
}

// ---------------------------------------------------------------------------
// K5: reduce K-split partials, add b_val (softmax rows sum to 1 -> bias passes
// through exactly), concat pos rows, zero-pad K to 576, cast bf16.
// ---------------------------------------------------------------------------
__global__ __launch_bounds__(256) void tok_reduce_cat(const float* __restrict__ part,
                                                      const float* __restrict__ b_val,
                                                      const float* __restrict__ pos,
                                                      unsigned short* __restrict__ tokcat) {
  int idx = blockIdx.x * 256 + threadIdx.x;
  if (idx >= NB * KCAT * LTOK) return;
  int l = idx & 63;
  int c = (idx >> 6) % KCAT;
  int n = idx / (KCAT * LTOK);
  float v;
  if (c < CT) {
    v = b_val[c];
#pragma unroll
    for (int ks = 0; ks < KSPLIT; ++ks)
      v += part[(((size_t)n * KSPLIT + ks) * CT + c) * 64 + l];
  } else if (c < CT + POSD) {
    v = pos[((size_t)n * POSD + (c - CT)) * LTOK + l];
  } else {
    v = 0.f;
  }
  tokcat[idx] = f2bf(v);
}

// ---------------------------------------------------------------------------
// K6: out[n] = w_tok (512x561, zero-padded to 576) x tokcat[n] + b_tok.
// grid(32), 512 thr; wave w owns rows [w*32, w*32+32).
// ---------------------------------------------------------------------------
__global__ __launch_bounds__(512) void tok_final_gemm(const unsigned short* __restrict__ tokcat,
                                                      const float* __restrict__ w_tok,
                                                      const float* __restrict__ b_tok,
                                                      float* __restrict__ out) {
  int n = blockIdx.x;
  int w = threadIdx.x >> 5;
  int row0 = w * 32;
  const unsigned short* bmat = tokcat + (size_t)n * KCAT * LTOK;
  v8f acc[8] = {};
  for (int k0 = 0; k0 < KCAT; k0 += 32) {
    v16bf a0 = load_a_wtok(w_tok, row0, k0);
    v16bf a1 = load_a_wtok(w_tok, row0 + 16, k0);
#pragma unroll
    for (int nt = 0; nt < 4; ++nt) {
      v16bf b = load_b_bf(bmat, k0, nt * 16, 64);
      acc[nt] = WMMA_BF16(a0, b, acc[nt]);
      acc[4 + nt] = WMMA_BF16(a1, b, acc[4 + nt]);
    }
  }
  int t = threadIdx.x & 31;
  int hi = (t < 16) ? 0 : 8, cl = t & 15;
#pragma unroll
  for (int mt = 0; mt < 2; ++mt)
#pragma unroll
    for (int nt = 0; nt < 4; ++nt)
#pragma unroll
      for (int r = 0; r < 8; ++r) {
        int d = row0 + mt * 16 + r + hi;
        out[((size_t)n * CT + d) * LTOK + nt * 16 + cl] = acc[mt * 4 + nt][r] + b_tok[d];
      }
}

// ---------------------------------------------------------------------------
extern "C" void kernel_launch(void* const* d_in, const int* in_sizes, int n_in,
                              void* d_out, int out_size, void* d_ws, size_t ws_size,
                              hipStream_t stream) {
  (void)in_sizes; (void)n_in; (void)out_size; (void)ws_size;
  const float* feature = (const float*)d_in[0];
  const float* w_tc    = (const float*)d_in[1];
  const float* b_tc    = (const float*)d_in[2];
  const float* w_val   = (const float*)d_in[3];
  const float* b_val   = (const float*)d_in[4];
  const float* w_ds3   = (const float*)d_in[5];
  const float* b_ds3   = (const float*)d_in[6];
  const float* w_ds1   = (const float*)d_in[7];
  const float* b_ds1   = (const float*)d_in[8];
  const float* w_pos   = (const float*)d_in[9];
  const float* b_pos   = (const float*)d_in[10];
  const float* w_tok   = (const float*)d_in[11];
  const float* b_tok   = (const float*)d_in[12];
  float* out = (float*)d_out;

  char* ws = (char*)d_ws;
  size_t o = 0;
  float* logits = (float*)(ws + o);          o += (size_t)NB * LTOK * HWD * 4;   // 25.7 MB
  float* stats  = (float*)(ws + o);          o += (size_t)NB * LTOK * 2 * 4;     // 16 KB
  unsigned short* tcb = (unsigned short*)(ws + o); o += (size_t)NB * LTOK * HWD * 2; // 12.8 MB
  float* pos    = (float*)(ws + o);          o += (size_t)NB * POSD * LTOK * 4;  // 0.4 MB
  float* part   = (float*)(ws + o);          o += (size_t)NB * KSPLIT * CT * 64 * 4; // 29.4 MB
  unsigned short* tokcat = (unsigned short*)(ws + o); // 2.4 MB

  tok_tc_gemm<<<dim3(HWD / 64, NB), 128, 0, stream>>>(feature, w_tc, b_tc, logits);
  tok_softmax<<<NB * LTOK, 256, 0, stream>>>(logits, tcb, stats);
  tok_pos<<<NB * LTOK, 64, 0, stream>>>(logits, stats, w_ds3, b_ds3, w_ds1, b_ds1,
                                        w_pos, b_pos, pos);
  tok_val_agg<<<dim3(KSPLIT, NB), 512, 0, stream>>>(feature, w_val, tcb, part);
  tok_reduce_cat<<<(NB * KCAT * LTOK + 255) / 256, 256, 0, stream>>>(part, b_val, pos, tokcat);
  tok_final_gemm<<<NB, 512, 0, stream>>>(tokcat, w_tok, b_tok, out);
}
